// Conv2dODENet_53266184404999
// MI455X (gfx1250) — compile-verified, matrix operand order checked by reading, not run on hardware
//
#include <hip/hip_runtime.h>
#include <hip/hip_bf16.h>

typedef _Float16 half_t;
typedef __attribute__((ext_vector_type(16))) _Float16 v16h;
typedef __attribute__((ext_vector_type(2)))  __fp16   v2fp16;
typedef __attribute__((ext_vector_type(8)))  float    v8f;
typedef __attribute__((ext_vector_type(4)))  unsigned v4u;
typedef __attribute__((ext_vector_type(8)))  int      v8i_t;
typedef __attribute__((ext_vector_type(4)))  int      v4i_t;

#define NPIX   32768       // B*H*W = 8*64*64
#define NELEM  2097152     // NPIX*64
#define NELEM4 524288

#if __has_builtin(__builtin_amdgcn_tensor_load_to_lds)
#define USE_TDM 1
#else
#define USE_TDM 0
#endif

// ---- Dormand-Prince tableau ----
#define C2f 0.2f
#define C3f 0.3f
#define C4f 0.8f
#define C5f (8.0f/9.0f)
#define A21f 0.2f
#define A31f (3.0f/40.0f)
#define A32f (9.0f/40.0f)
#define A41f (44.0f/45.0f)
#define A42f (-56.0f/15.0f)
#define A43f (32.0f/9.0f)
#define A51f (19372.0f/6561.0f)
#define A52f (-25360.0f/2187.0f)
#define A53f (64448.0f/6561.0f)
#define A54f (-212.0f/729.0f)
#define A61f (9017.0f/3168.0f)
#define A62f (-355.0f/33.0f)
#define A63f (46732.0f/5247.0f)
#define A64f (49.0f/176.0f)
#define A65f (-5103.0f/18656.0f)
#define B1f (35.0f/384.0f)
#define B3f (500.0f/1113.0f)
#define B4f (125.0f/192.0f)
#define B5f (-2187.0f/6784.0f)
#define B6f (11.0f/84.0f)
#define E1f (71.0f/57600.0f)
#define E3f (-71.0f/16695.0f)
#define E4f (71.0f/1920.0f)
#define E5f (-17253.0f/339200.0f)
#define E6f (22.0f/525.0f)
#define E7f (-1.0f/40.0f)
#define TOLf 1e-3f

__device__ inline unsigned pack2(float a, float b) {
    union { v2fp16 h; unsigned u; } p;
    p.h = __builtin_amdgcn_cvt_pkrtz(a, b);   // one v_cvt_pk_rtz_f16_f32
    return p.u;
}

// Fragment loader matching CDNA5 16-bit A/B layout:
// lanes 0-15  : halves K {0..7} and {16..23} of row (lane&15)
// lanes 16-31 : halves K {8..15} and {24..31}
// LDS layout: row stride 20 uints (40 halves, padded against bank conflicts)
__device__ inline v16h load_frag(const unsigned* base, int row, int lane) {
    const int sel = (lane >> 4) & 1;
    union { v16h h; uint4 u[2]; } f;
    f.u[0] = *(const uint4*)(base + row * 20 + sel * 4);
    f.u[1] = *(const uint4*)(base + row * 20 + 8 + sel * 4);
    return f.h;
}

__device__ inline v8f wmma16(v16h a, v16h b, v8f c) {
    return __builtin_amdgcn_wmma_f32_16x16x32_f16(
        /*neg_a=*/false, a, /*neg_b=*/false, b,
        /*c_mod=*/(short)0, c, /*reuse_a=*/false, /*reuse_b=*/false);
}

#if USE_TDM
// TDM descriptor: 2D tile [NOUT rows][32 halves] out of tensor row-stride KTOT
// halves, landing in LDS with a 4-DWORD pad after every 16 DWORDs -> exactly
// the padded 40-half row stride the fragment loader expects.
template<int KTOT, int NOUT>
__device__ inline void tdm_load_B(const half_t* gsrc, unsigned lds_byte) {
    const unsigned long long ga = (unsigned long long)gsrc;
    v4u g0;
    g0[0] = 1u;                                             // count=1 (valid D#)
    g0[1] = lds_byte;                                       // lds_addr
    g0[2] = (unsigned)ga;                                   // global_addr[31:0]
    g0[3] = (unsigned)((ga >> 32) & 0x1FFFFFFu) | (2u << 30); // addr[56:32], type=2
    v8i_t g1;
    g1[0] = (int)((1u << 16)      // data_size = 2 bytes
                | (1u << 20)      // pad_enable
                | (3u << 22)      // pad_interval: 16 DWORDs (64B = one tile row)
                | (3u << 25));    // pad_amount: 4 DWORDs (16B pad -> stride 80B)
    g1[1] = (int)(((unsigned)KTOT & 0xFFFFu) << 16);        // tensor_dim0 lo
    g1[2] = (int)((((unsigned)KTOT >> 16) & 0xFFFFu)
                | (((unsigned)NOUT & 0xFFFFu) << 16));      // tensor_dim0 hi | dim1 lo
    g1[3] = (int)(32u << 16);                               // dim1 hi=0 | tile_dim0=32
    g1[4] = NOUT;                                           // tile_dim1 (tile_dim2=0)
    g1[5] = KTOT;                                           // tensor_dim0_stride lo
    g1[6] = 0;                                              // stride hi | dim1_stride lo
    g1[7] = 0;
    const v4i_t z4 = {0, 0, 0, 0};                          // 2D: groups 2/3 unused
    const v8i_t z8 = {0, 0, 0, 0, 0, 0, 0, 0};
    __builtin_amdgcn_tensor_load_to_lds(g0, g1, z4, z4, z8, 0);
}
#endif

// Implicit-GEMM conv, A=[NPIX][CIN] f32 activations, BT=[NOUT][TAPS*CIN] f16.
// TMODE 0: plain bias (head).  TMODE 1: bias + t*tw[n] (1x1 with time channel).
// TMODE 2: bias + t*sum_{valid taps} w2[tap][0][n] (3x3 SAME with time channel).
template<int CIN, int TAPS, int NOUT, int NSTORE, bool RELU, int TMODE>
__global__ __launch_bounds__(256) void conv_gemm(
    const float* __restrict__ A,
    const half_t* __restrict__ BT,
    const float* __restrict__ bias,
    const float* __restrict__ tw,
    const float* __restrict__ state,
    float cstage,
    float* __restrict__ out)
{
    constexpr int KTOT = TAPS * CIN;
    constexpr int WN = (NOUT >= 128) ? 2 : 1;
    constexpr int WM = 8 / WN;
    constexpr int TM = 128 / WM;
    constexpr int TN = NOUT / WN;
    constexpr int MI = TM / 16;
    constexpr int NJ = TN / 16;

    __shared__ unsigned Asu[128 * 20];
    __shared__ unsigned Bsu[NOUT * 20];

    const int tid  = threadIdx.x;
    const int lane = tid & 31;
    const int wid  = tid >> 5;
    const int wm   = wid % WM;
    const int wn   = wid / WM;
    const int m0   = blockIdx.x * 128;

    float tst = 0.0f;
    if (TMODE != 0) tst = state[0] + state[2] * cstage;

    v8f acc[MI][NJ];
#pragma unroll
    for (int i = 0; i < MI; ++i)
#pragma unroll
        for (int j = 0; j < NJ; ++j) acc[i][j] = (v8f){0,0,0,0,0,0,0,0};

#if !USE_TDM
    const unsigned* BTu = (const unsigned*)BT;
#endif

    for (int kt = 0; kt < KTOT; kt += 32) {
        __syncthreads();
#if USE_TDM
        // ---- B tile via Tensor Data Mover (wave 0 only; EXEC-independent DMA)
        if (wid == 0)
            tdm_load_B<KTOT, NOUT>(BT + kt,
                (unsigned)(unsigned long long)(const void*)Bsu);
#endif
        // ---- stage A tile (128 rows x 32 halves), f32 -> f16 ----
        {
            const int r   = tid >> 1;
            const int seg = tid & 1;
            const int p   = m0 + r;
            const float* src = nullptr;
            bool valid = true;
            if (TAPS == 1) {
                src = A + (size_t)p * CIN + kt + seg * 16;
                if (kt + 32 < KTOT) __builtin_prefetch(src + 32, 0, 0);
            } else {
                const int tap = kt / CIN;
                const int c0  = kt % CIN;
                const int dy  = tap / 3, dx = tap % 3;
                const int b   = p >> 12;
                const int rem = p & 4095;
                const int yy  = (rem >> 6) + dy - 1;
                const int xx  = (rem & 63) + dx - 1;
                valid = ((unsigned)yy < 64u) && ((unsigned)xx < 64u);
                src = A + ((size_t)((b << 12) + (yy << 6) + xx)) * CIN + c0 + seg * 16;
                if (valid) __builtin_prefetch(src + 16, 0, 0);
            }
            float4 f0 = {0,0,0,0}, f1 = {0,0,0,0}, f2 = {0,0,0,0}, f3 = {0,0,0,0};
            if (valid) {
                const float4* s4 = (const float4*)src;
                f0 = s4[0]; f1 = s4[1]; f2 = s4[2]; f3 = s4[3];
            }
            unsigned* dst = Asu + r * 20 + seg * 8;
            dst[0] = pack2(f0.x, f0.y); dst[1] = pack2(f0.z, f0.w);
            dst[2] = pack2(f1.x, f1.y); dst[3] = pack2(f1.z, f1.w);
            dst[4] = pack2(f2.x, f2.y); dst[5] = pack2(f2.z, f2.w);
            dst[6] = pack2(f3.x, f3.y); dst[7] = pack2(f3.z, f3.w);
        }
#if USE_TDM
        if (wid == 0) __builtin_amdgcn_s_wait_tensorcnt(0);
#else
        // ---- fallback: stage B tile with vector copies ----
        for (int idx = tid; idx < NOUT * 8; idx += 256) {
            const int n = idx >> 3, w = idx & 7;
            Bsu[n * 20 + w] = BTu[(size_t)n * (KTOT / 2) + (kt >> 1) + w];
        }
#endif
        __syncthreads();

        v16h af[MI], bf[NJ];
#pragma unroll
        for (int i = 0; i < MI; ++i)
            af[i] = load_frag(Asu, wm * TM + i * 16 + (lane & 15), lane);
#pragma unroll
        for (int j = 0; j < NJ; ++j)
            bf[j] = load_frag(Bsu, wn * TN + j * 16 + (lane & 15), lane);
#pragma unroll
        for (int i = 0; i < MI; ++i)
#pragma unroll
            for (int j = 0; j < NJ; ++j)
                acc[i][j] = wmma16(af[i], bf[j], acc[i][j]);
    }

    // ---- epilogue ----
    const float* Ls = (const float*)Asu;
    if (TMODE == 2) {
        __syncthreads();
        float* Lw = (float*)Asu;                 // reuse A LDS for 9x128 time-weight table
        for (int idx = tid; idx < 9 * 128; idx += 256) {
            const int tap = idx >> 7, f = idx & 127;
            Lw[idx] = tw[(size_t)tap * (129 * 128) + f];   // w2[dy][dx][0][f]
        }
        __syncthreads();
    }

#pragma unroll
    for (int i = 0; i < MI; ++i) {
#pragma unroll
        for (int j = 0; j < NJ; ++j) {
            const int n = wn * TN + j * 16 + (lane & 15);
            float bn;
            if (TMODE == 0)      bn = (n < NSTORE) ? bias[n] : 0.0f;
            else if (TMODE == 1) bn = bias[n] + tst * tw[n];
            else                 bn = bias[n];
#pragma unroll
            for (int rr = 0; rr < 8; ++rr) {
                const int m  = wm * TM + i * 16 + ((lane >> 4) << 3) + rr;
                const int mg = m0 + m;
                float v = acc[i][j][rr] + bn;
                if (TMODE == 2) {
                    const int rem = mg & 4095;
                    const int yy = rem >> 6, xx = rem & 63;
                    float ts = 0.0f;
#pragma unroll
                    for (int dy = 0; dy < 3; ++dy)
#pragma unroll
                        for (int dx = 0; dx < 3; ++dx)
                            if ((unsigned)(yy + dy - 1) < 64u &&
                                (unsigned)(xx + dx - 1) < 64u)
                                ts += Ls[(dy * 3 + dx) * 128 + n];
                    v += tst * ts;
                }
                if (RELU) v = fmaxf(v, 0.0f);
                if (NSTORE == NOUT || n < NSTORE)
                    out[(size_t)mg * NSTORE + n] = v;
            }
        }
    }
}

// dst = y + hs * (c1*p1 + ... + cNK*pNK)
template<int NK>
__global__ __launch_bounds__(256) void combine_kernel(
    float* __restrict__ dst, const float* __restrict__ ybase,
    const float* __restrict__ state,
    const float* __restrict__ p1, const float* __restrict__ p2,
    const float* __restrict__ p3, const float* __restrict__ p4,
    const float* __restrict__ p5,
    float c1, float c2, float c3, float c4, float c5)
{
    const float hs = state[2];
    for (size_t i = (size_t)blockIdx.x * blockDim.x + threadIdx.x;
         i < NELEM4; i += (size_t)gridDim.x * blockDim.x) {
        const float4 yv = ((const float4*)ybase)[i];
        float sx = 0, sy = 0, sz = 0, sw = 0;
        if constexpr (NK >= 1) { float4 v = ((const float4*)p1)[i]; sx += c1*v.x; sy += c1*v.y; sz += c1*v.z; sw += c1*v.w; }
        if constexpr (NK >= 2) { float4 v = ((const float4*)p2)[i]; sx += c2*v.x; sy += c2*v.y; sz += c2*v.z; sw += c2*v.w; }
        if constexpr (NK >= 3) { float4 v = ((const float4*)p3)[i]; sx += c3*v.x; sy += c3*v.y; sz += c3*v.z; sw += c3*v.w; }
        if constexpr (NK >= 4) { float4 v = ((const float4*)p4)[i]; sx += c4*v.x; sy += c4*v.y; sz += c4*v.z; sw += c4*v.w; }
        if constexpr (NK >= 5) { float4 v = ((const float4*)p5)[i]; sx += c5*v.x; sy += c5*v.y; sz += c5*v.z; sw += c5*v.w; }
        float4 o;
        o.x = yv.x + hs * sx; o.y = yv.y + hs * sy;
        o.z = yv.z + hs * sz; o.w = yv.w + hs * sw;
        ((float4*)dst)[i] = o;
    }
}

__global__ __launch_bounds__(256) void err_reduce_kernel(
    const float* __restrict__ y,  const float* __restrict__ y5,
    const float* __restrict__ k1, const float* __restrict__ k3,
    const float* __restrict__ k4, const float* __restrict__ k5,
    const float* __restrict__ k6, const float* __restrict__ k7,
    float* __restrict__ state)
{
    const float hs = state[2];
    float part = 0.0f;
    for (size_t i = (size_t)blockIdx.x * blockDim.x + threadIdx.x;
         i < NELEM4; i += (size_t)gridDim.x * blockDim.x) {
        const float4 a1 = ((const float4*)k1)[i];
        const float4 a3 = ((const float4*)k3)[i];
        const float4 a4 = ((const float4*)k4)[i];
        const float4 a5 = ((const float4*)k5)[i];
        const float4 a6 = ((const float4*)k6)[i];
        const float4 a7 = ((const float4*)k7)[i];
        const float4 yv = ((const float4*)y)[i];
        const float4 y5v = ((const float4*)y5)[i];
#pragma unroll
        for (int c = 0; c < 4; ++c) {
            const float e = hs * (E1f * (&a1.x)[c] + E3f * (&a3.x)[c] + E4f * (&a4.x)[c]
                                + E5f * (&a5.x)[c] + E6f * (&a6.x)[c] + E7f * (&a7.x)[c]);
            const float sc = TOLf + TOLf * fmaxf(fabsf((&yv.x)[c]), fabsf((&y5v.x)[c]));
            const float r = e / sc;
            part += r * r;
        }
    }
    __shared__ float red[256];
    red[threadIdx.x] = part;
    __syncthreads();
    for (int s = 128; s > 0; s >>= 1) {
        if (threadIdx.x < s) red[threadIdx.x] += red[threadIdx.x + s];
        __syncthreads();
    }
    if (threadIdx.x == 0) atomicAdd(&state[4], red[0]);
}

__global__ void init_state_kernel(float* s) {
    s[0] = 0.0f;   // t
    s[1] = 0.1f;   // h
    s[2] = 0.0f; s[3] = 0.0f; s[4] = 0.0f; s[5] = 0.0f; s[6] = 0.0f; s[7] = 0.0f;
}

__global__ void step_begin_kernel(float* s) {
    const float t = s[0], h = s[1];
    s[2] = fminf(h, 1.0f - t);          // hs
    s[4] = 0.0f;                        // error accumulator
    s[5] = (t >= 1.0f) ? 1.0f : 0.0f;   // done
}

__global__ void step_end_kernel(float* s) {
    const float t = s[0], h = s[1], hs = s[2];
    const float en = sqrtf(s[4] / (float)NELEM);
    const bool done = (s[5] != 0.0f);
    const bool accept = (en <= 1.0f) && !done;
    s[0] = accept ? (t + hs) : t;
    const float ens = fmaxf(en, 1e-8f);
    float fac = 0.9f * powf(ens, -0.2f);
    fac = fminf(fmaxf(fac, 0.2f), 10.0f);
    s[1] = done ? h : fmaxf(hs * fac, 1e-4f);
    s[3] = accept ? 1.0f : 0.0f;
}

__global__ __launch_bounds__(256) void y_select_kernel(
    float* __restrict__ y, const float* __restrict__ y5,
    const float* __restrict__ state)
{
    if (state[3] == 0.0f) return;       // uniform: step rejected, keep y
    for (size_t i = (size_t)blockIdx.x * blockDim.x + threadIdx.x;
         i < NELEM4; i += (size_t)gridDim.x * blockDim.x)
        ((float4*)y)[i] = ((const float4*)y5)[i];
}

// HWIO f32 -> [ndst][taps*cin_used] f16 (transposed, time channel skipped)
__global__ __launch_bounds__(256) void prep_weights_kernel(
    const float* __restrict__ w, half_t* __restrict__ outT,
    int taps, int cin_total, int skip, int cin_used, int nsrc, int ndst)
{
    const int KT = taps * cin_used;
    const int total = ndst * KT;
    for (int idx = blockIdx.x * blockDim.x + threadIdx.x; idx < total;
         idx += gridDim.x * blockDim.x) {
        const int n = idx / KT, k = idx % KT;
        const int tap = k / cin_used, c = k % cin_used;
        const float v = (n < nsrc)
            ? w[((size_t)tap * cin_total + skip + c) * (size_t)nsrc + n] : 0.0f;
        outT[(size_t)n * KT + k] = (half_t)v;
    }
}

extern "C" void kernel_launch(void* const* d_in, const int* in_sizes, int n_in,
                              void* d_out, int out_size, void* d_ws, size_t ws_size,
                              hipStream_t stream) {
    (void)in_sizes; (void)n_in; (void)out_size; (void)ws_size;
    const float* x  = (const float*)d_in[0];
    const float* w1 = (const float*)d_in[1];   // [1,1,65,128]
    const float* b1 = (const float*)d_in[2];
    const float* w2 = (const float*)d_in[3];   // [3,3,129,128]
    const float* b2 = (const float*)d_in[4];
    const float* w3 = (const float*)d_in[5];   // [1,1,129,64]
    const float* b3 = (const float*)d_in[6];
    const float* wo = (const float*)d_in[7];   // [1,1,64,10]
    const float* bo = (const float*)d_in[8];
    float* out = (float*)d_out;

    // ---- workspace layout ----
    float* base = (float*)d_ws;
    size_t off = 0;
    auto alloc = [&](size_t n) { float* p = base + off; off += n; return p; };
    float* y      = alloc(NELEM);
    float* kbuf[7];
    for (int i = 0; i < 7; ++i) kbuf[i] = alloc(NELEM);
    float* ystage = alloc(NELEM);
    float* y5     = alloc(NELEM);
    float* h1     = alloc((size_t)NPIX * 128);
    float* h2     = alloc((size_t)NPIX * 128);
    float* state  = alloc(16);
    half_t* w1T = (half_t*)(base + off); off += (128 * 64) / 2;
    half_t* w2T = (half_t*)(base + off); off += (128 * 1152) / 2;
    half_t* w3T = (half_t*)(base + off); off += (64 * 128) / 2;
    half_t* woT = (half_t*)(base + off); off += (16 * 64) / 2;

    // ---- one-time (per launch) prep ----
    (void)hipMemcpyAsync(y, x, (size_t)NELEM * sizeof(float),
                         hipMemcpyDeviceToDevice, stream);
    init_state_kernel<<<1, 1, 0, stream>>>(state);
    prep_weights_kernel<<<(128*64   + 255)/256, 256, 0, stream>>>(w1, w1T, 1,  65, 1,  64, 128, 128);
    prep_weights_kernel<<<(128*1152 + 255)/256, 256, 0, stream>>>(w2, w2T, 9, 129, 1, 128, 128, 128);
    prep_weights_kernel<<<(64*128   + 255)/256, 256, 0, stream>>>(w3, w3T, 1, 129, 1, 128,  64,  64);
    prep_weights_kernel<<<(16*64    + 255)/256, 256, 0, stream>>>(wo, woT, 1,  64, 0,  64,  10,  16);

    const dim3 gG(256), gB(256);     // GEMM grid: 32768/128 blocks
    const dim3 eG(512), eB(256);     // elementwise grid

    auto feval = [&](const float* yin, float* kout, float cstage) {
        conv_gemm<64, 1, 128, 128, true, 1><<<gG, gB, 0, stream>>>(
            yin, w1T, b1, w1 /* tw[n]=w1[0][n] */, state, cstage, h1);
        conv_gemm<128, 9, 128, 128, true, 2><<<gG, gB, 0, stream>>>(
            h1, w2T, b2, w2 /* tap base */, state, cstage, h2);
        conv_gemm<128, 1, 64, 64, false, 1><<<gG, gB, 0, stream>>>(
            h2, w3T, b3, w3 /* tw[n]=w3[0][n] */, state, cstage, kout);
    };

    for (int it = 0; it < 32; ++it) {
        step_begin_kernel<<<1, 1, 0, stream>>>(state);
        feval(y, kbuf[0], 0.0f);                                    // k1 at t
        combine_kernel<1><<<eG, eB, 0, stream>>>(ystage, y, state,
            kbuf[0], nullptr, nullptr, nullptr, nullptr,
            A21f, 0, 0, 0, 0);
        feval(ystage, kbuf[1], C2f);                                // k2
        combine_kernel<2><<<eG, eB, 0, stream>>>(ystage, y, state,
            kbuf[0], kbuf[1], nullptr, nullptr, nullptr,
            A31f, A32f, 0, 0, 0);
        feval(ystage, kbuf[2], C3f);                                // k3
        combine_kernel<3><<<eG, eB, 0, stream>>>(ystage, y, state,
            kbuf[0], kbuf[1], kbuf[2], nullptr, nullptr,
            A41f, A42f, A43f, 0, 0);
        feval(ystage, kbuf[3], C4f);                                // k4
        combine_kernel<4><<<eG, eB, 0, stream>>>(ystage, y, state,
            kbuf[0], kbuf[1], kbuf[2], kbuf[3], nullptr,
            A51f, A52f, A53f, A54f, 0);
        feval(ystage, kbuf[4], C5f);                                // k5
        combine_kernel<5><<<eG, eB, 0, stream>>>(ystage, y, state,
            kbuf[0], kbuf[1], kbuf[2], kbuf[3], kbuf[4],
            A61f, A62f, A63f, A64f, A65f);
        feval(ystage, kbuf[5], 1.0f);                               // k6 at t+hs
        combine_kernel<5><<<eG, eB, 0, stream>>>(y5, y, state,
            kbuf[0], kbuf[2], kbuf[3], kbuf[4], kbuf[5],
            B1f, B3f, B4f, B5f, B6f);                               // y5
        feval(y5, kbuf[6], 1.0f);                                   // k7 (FSAL)
        err_reduce_kernel<<<eG, eB, 0, stream>>>(y, y5,
            kbuf[0], kbuf[2], kbuf[3], kbuf[4], kbuf[5], kbuf[6], state);
        step_end_kernel<<<1, 1, 0, stream>>>(state);
        y_select_kernel<<<eG, eB, 0, stream>>>(y, y5, state);
    }

    // output head: 1x1 conv 64 -> 10 (N padded to 16, store 10)
    conv_gemm<64, 1, 16, 10, false, 0><<<gG, gB, 0, stream>>>(
        y, woT, bo, nullptr, nullptr, 0.0f, out);
}
